// MultiHeadAttention_16492674417334
// MI455X (gfx1250) — matrix operand order repaired and checked by reading, not compiled, not run on hardware
//
#include <hip/hip_runtime.h>

// ---------------------------------------------------------------------------
// MHA forward for gfx1250 (MI455X): f16 WMMA + fp32 accumulate everywhere,
// FlashAttention-style fused softmax (no S x S matrix in HBM), and
// Tensor-Data-Mover (tensor_load_to_lds) double-buffered tile staging so the
// vector pipes stay free for v_wmma issue.
// ---------------------------------------------------------------------------

typedef __attribute__((ext_vector_type(16))) _Float16 v16h;
typedef __attribute__((ext_vector_type(8)))  _Float16 v8h;
typedef __attribute__((ext_vector_type(4)))  _Float16 v4h;
typedef __attribute__((ext_vector_type(8)))  float    v8f;
typedef __attribute__((ext_vector_type(4)))  unsigned int u32x4;
typedef __attribute__((ext_vector_type(8)))  unsigned int u32x8;

#define WMMA_F16(a, b, c) \
  __builtin_amdgcn_wmma_f32_16x16x32_f16(false, (a), false, (b), (short)0, (c), false, false)

#define NB 2
#define NS 2048
#define ND 1024
#define NH 16
#define NHD 64

union FragU { v16h v; struct { v8h lo; v8h hi; } p; };

// ---------------------------------------------------------------------------
// Tensor Data Mover: async 2D tile load (global -> LDS), f16 elements.
// Builds the D# per CDNA5 ISA and issues tensor_load_to_lds
// with the 2-SGPR-group form (tile <= 2D, VADDR2/3 = NULL).
//   group0: [0]=count=1, [1]=lds byte addr, [2..3]=57b global addr | type=2
//   group1: data_size=1 (2B), tensor_dim0/1, tile_dim0/1, dim0 stride
// Completion is tracked with TENSORcnt (s_wait_tensorcnt).
// ---------------------------------------------------------------------------
__device__ __forceinline__ void tdm_load_2d_f16(unsigned lds_addr,
                                                const _Float16* gptr,
                                                unsigned tensor_d0,  // elems/row
                                                unsigned tensor_d1,  // rows
                                                unsigned tile_d0,    // elems/row
                                                unsigned tile_d1,    // rows
                                                unsigned row_stride) // elems
{
  const unsigned long long ga = (unsigned long long)(size_t)gptr;
  u32x4 g0;
  g0[0] = 1u;                                             // count=1
  g0[1] = lds_addr;                                       // lds_addr
  g0[2] = (unsigned)(ga & 0xFFFFFFFFu);                   // global_addr[31:0]
  g0[3] = (unsigned)((ga >> 32) & 0x01FFFFFFu) | (2u << 30); // addr[56:32]|type=2
  u32x8 g1;
  g1[0] = 1u << 16;                                       // data_size=1 (2B)
  g1[1] = (tensor_d0 & 0xFFFFu) << 16;                    // tensor_dim0[15:0]
  g1[2] = (tensor_d0 >> 16) | ((tensor_d1 & 0xFFFFu) << 16);
  g1[3] = (tensor_d1 >> 16) | (tile_d0 << 16);            // tile_dim0
  g1[4] = tile_d1;                                        // tile_dim1, tile_dim2=0
  g1[5] = row_stride;                                     // dim0_stride[31:0]
  g1[6] = 0u;
  g1[7] = 0u;
  asm volatile("tensor_load_to_lds %0, %1" :: "s"(g0), "s"(g1) : "memory");
}

__device__ __forceinline__ unsigned lds_offset_of(const void* p) {
  return (unsigned)(size_t)p;   // low 32 bits of generic addr == LDS byte offset
}

// ---------------------------------------------------------------------------
// Cast fp32 -> f16 (x and weight matrices), 4 elements / thread.
// ---------------------------------------------------------------------------
__global__ __launch_bounds__(256)
void cast_f32_to_f16(const float* __restrict__ src, _Float16* __restrict__ dst, int n) {
  int i = (blockIdx.x * 256 + threadIdx.x) * 4;
  if (i + 3 < n) {
    float4 f = *reinterpret_cast<const float4*>(src + i);
    v4h h;
    h[0] = (_Float16)f.x; h[1] = (_Float16)f.y;
    h[2] = (_Float16)f.z; h[3] = (_Float16)f.w;
    *reinterpret_cast<v4h*>(dst + i) = h;
  }
}

// ---------------------------------------------------------------------------
// C[M,N] = A[M,K] * W[N,K]^T + bias.   M=4096 (rowsA), N=K=1024.
// Block tile 128x128, K-panel 64, double-buffered via TDM; 8 waves (wave32),
// each wave owns a 32x64 sub-tile (8 fp32 WMMA accumulators).
// mode 0: store f16 rearranged to [B,H,S,HD];  mode 1: store fp32 [M,N].
// W row-major [N][K] is already the WMMA B-fragment (Bt[n][k]) layout.
// ---------------------------------------------------------------------------
__global__ __launch_bounds__(256)
void gemm_xwT(const _Float16* __restrict__ A, const _Float16* __restrict__ W,
              const float* __restrict__ bias, _Float16* __restrict__ out_h,
              float* __restrict__ out_f, int rowsA, int mode) {
  __shared__ __align__(128) _Float16 As[2][128 * 64];
  __shared__ __align__(128) _Float16 Bs[2][128 * 64];

  const int tid   = threadIdx.x;
  const int lane  = tid & 31;
  const int wave  = tid >> 5;     // 0..7
  const int wm    = wave & 3;     // 4 waves along M
  const int wn    = wave >> 2;    // 2 waves along N
  const int m0    = blockIdx.x * 128;
  const int n0    = blockIdx.y * 128;
  const int lhalf = lane & 15;
  const int hsel  = lane >> 4;    // 0: lanes 0-15, 1: lanes 16-31

  v8f acc[2][4];
  const v8f vzero = {};
  for (int mt = 0; mt < 2; ++mt)
    for (int nt = 0; nt < 4; ++nt) acc[mt][nt] = vzero;

  // Prologue: DMA first K-panel into buffer 0 (one wave issues for the block).
  if (wave == 0) {
    tdm_load_2d_f16(lds_offset_of(&As[0][0]), &A[(size_t)m0 * ND],
                    ND, (unsigned)rowsA, 64, 128, ND);
    tdm_load_2d_f16(lds_offset_of(&Bs[0][0]), &W[(size_t)n0 * ND],
                    ND, ND, 64, 128, ND);
  }

  for (int k0 = 0; k0 < ND; k0 += 64) {
    const int cb = (k0 >> 6) & 1;
    if (wave == 0) __builtin_amdgcn_s_wait_tensorcnt(0);  // panel cb resident
    __syncthreads();
    // Kick the DMA for the next panel while everyone computes this one.
    if (wave == 0 && (k0 + 64) < ND) {
      tdm_load_2d_f16(lds_offset_of(&As[cb ^ 1][0]), &A[(size_t)m0 * ND + k0 + 64],
                      ND, (unsigned)rowsA, 64, 128, ND);
      tdm_load_2d_f16(lds_offset_of(&Bs[cb ^ 1][0]), &W[(size_t)n0 * ND + k0 + 64],
                      ND, ND, 64, 128, ND);
    }

    const _Float16* __restrict__ at = &As[cb][0];
    const _Float16* __restrict__ bt = &Bs[cb][0];
    for (int kk = 0; kk < 64; kk += 32) {
      v16h afrag[2];
      for (int mt = 0; mt < 2; ++mt) {
        const _Float16* ap = &at[(wm * 32 + mt * 16 + lhalf) * 64 + kk + hsel * 16];
        FragU u;
        u.p.lo = *reinterpret_cast<const v8h*>(ap);        // K = kk+hs*16 .. +7
        u.p.hi = *reinterpret_cast<const v8h*>(ap + 8);    // K = kk+hs*16+8 ..
        afrag[mt] = u.v;
      }
      for (int nt = 0; nt < 4; ++nt) {
        const _Float16* bp = &bt[(wn * 64 + nt * 16 + lhalf) * 64 + kk + hsel * 16];
        v16h bfrag = *reinterpret_cast<const v16h*>(bp);   // 16 contiguous K
        for (int mt = 0; mt < 2; ++mt)
          acc[mt][nt] = WMMA_F16(afrag[mt], bfrag, acc[mt][nt]);
      }
    }
    __syncthreads();  // everyone done with panel cb before DMA of k0+128 lands
  }

  // Epilogue. C layout: vgpr r -> row (hsel*8 + r), lane%16 -> col.
  for (int mt = 0; mt < 2; ++mt) {
    for (int nt = 0; nt < 4; ++nt) {
      const int gn = n0 + wn * 64 + nt * 16 + lhalf;
      const float bv = bias[gn];
      const int rowbase = m0 + wm * 32 + mt * 16 + hsel * 8;
      for (int r = 0; r < 8; ++r) {
        const int gm = rowbase + r;
        const float v = acc[mt][nt][r] + bv;
        if (mode == 0) {
          const int bi = gm >> 11, s = gm & (NS - 1);
          const int h = gn >> 6, d = gn & (NHD - 1);
          out_h[(((size_t)bi * NH + h) * NS + s) * NHD + d] = (_Float16)v;
        } else {
          out_f[(size_t)gm * ND + gn] = v;
        }
      }
    }
  }
}

// ---------------------------------------------------------------------------
// Fused causal+padded attention, online softmax. One block = one (b,h) pair
// and 64 query rows; 4 waves, each wave owns 16 query rows. Q/K tiles are
// DMA'd by the TDM; V is transposed into LDS by the vector pipe (overlapping
// the in-flight K DMA). Scores and P*V both via v_wmma_f32_16x16x32_f16.
// ---------------------------------------------------------------------------
__global__ __launch_bounds__(128)
void flash_attn(const _Float16* __restrict__ Qh, const _Float16* __restrict__ Kh,
                const _Float16* __restrict__ Vh, const unsigned char* __restrict__ pad,
                _Float16* __restrict__ Oh) {
  __shared__ __align__(128) _Float16 Qs[64 * 64];
  __shared__ __align__(128) _Float16 Ks[64 * 64];
  __shared__ __align__(128) _Float16 Vt[64 * 64];      // Vt[d][key]
  __shared__ __align__(128) _Float16 Ps[4][16 * 64];   // per-wave probs (A layout)

  const int tid   = threadIdx.x;
  const int lane  = tid & 31;
  const int wave  = tid >> 5;      // 0..3 -> query rows wave*16 .. +15
  const int lhalf = lane & 15;
  const int hsel  = lane >> 4;

  const int qt = blockIdx.x;       // query tile (0..31)
  const int bh = blockIdx.y;       // 0..31
  const int b  = bh >> 4;
  const int h  = bh & (NH - 1);
  const size_t base = (size_t)bh * NS * NHD;

  // DMA the Q tile (64x64 f16) once; first in-loop wait covers it.
  if (wave == 0)
    tdm_load_2d_f16(lds_offset_of(&Qs[0]), &Qh[base + (size_t)qt * 64 * NHD],
                    NHD, NS, 64, 64, NHD);

  float m_i[8], l_i[8];
  v8f o[4];
  const v8f vzero = {};
  for (int r = 0; r < 8; ++r) { m_i[r] = -1.0e30f; l_i[r] = 0.0f; }
  for (int nt = 0; nt < 4; ++nt) o[nt] = vzero;

  const float scale = 0.125f;  // 1/sqrt(64)

  for (int j = 0; j <= qt; ++j) {   // causal: only key tiles <= query tile
    __syncthreads();                // prev iteration's reads of Ks/Vt done
    if (wave == 0)                  // DMA K tile (straight 64x64 copy)
      tdm_load_2d_f16(lds_offset_of(&Ks[0]), &Kh[base + (size_t)j * 64 * NHD],
                      NHD, NS, 64, 64, NHD);
    {  // V tile, transposed into Vt[d][key] (overlaps the K DMA)
      const int r = tid & 63;
      const int d0 = (tid >> 6) * 32;
      const _Float16* vp = &Vh[base + (size_t)(j * 64 + r) * NHD + d0];
      for (int cc = 0; cc < 4; ++cc) {
        v8h vv = *reinterpret_cast<const v8h*>(vp + cc * 8);
        for (int d = 0; d < 8; ++d) Vt[(d0 + cc * 8 + d) * 64 + r] = vv[d];
      }
    }
    if (wave == 0) __builtin_amdgcn_s_wait_tensorcnt(0);
    __syncthreads();

    // ---- scores = Q * K^T (M=16 queries, N=64 keys, K=HD=64) ----
    v8f sc[4];
    for (int nt = 0; nt < 4; ++nt) sc[nt] = vzero;
    for (int kk = 0; kk < 64; kk += 32) {
      const _Float16* ap = &Qs[(wave * 16 + lhalf) * 64 + kk + hsel * 16];
      FragU ua;
      ua.p.lo = *reinterpret_cast<const v8h*>(ap);
      ua.p.hi = *reinterpret_cast<const v8h*>(ap + 8);
      for (int nt = 0; nt < 4; ++nt) {
        const _Float16* bp = &Ks[(nt * 16 + lhalf) * 64 + kk + hsel * 16];
        v16h bfrag = *reinterpret_cast<const v16h*>(bp);
        sc[nt] = WMMA_F16(ua.v, bfrag, sc[nt]);
      }
    }

    // ---- mask + scale + online softmax ----
    float rowmax[8];
    for (int r = 0; r < 8; ++r) rowmax[r] = -1.0e30f;
    const int qrow0 = qt * 64 + wave * 16 + hsel * 8;
    for (int nt = 0; nt < 4; ++nt) {
      const int kg = j * 64 + nt * 16 + lhalf;
      const bool padk = pad[b * NS + kg] != 0;
      for (int r = 0; r < 8; ++r) {
        float v = sc[nt][r] * scale;
        if (padk || kg > qrow0 + r) v = -1.0e9f;
        sc[nt][r] = v;
        rowmax[r] = fmaxf(rowmax[r], v);
      }
    }
    // row lives on 16 lanes of one half-wave -> xor-reduce within width 16
    for (int off = 8; off >= 1; off >>= 1)
      for (int r = 0; r < 8; ++r)
        rowmax[r] = fmaxf(rowmax[r], __shfl_xor(rowmax[r], off, 16));

    float alpha[8], rowsum[8];
    for (int r = 0; r < 8; ++r) {
      const float mnew = fmaxf(m_i[r], rowmax[r]);
      alpha[r] = __expf(m_i[r] - mnew);
      m_i[r] = mnew;
      rowsum[r] = 0.0f;
    }

    // probs -> LDS (converts C layout to A layout for the P*V WMMA)
    _Float16* pw = &Ps[wave][0];
    for (int nt = 0; nt < 4; ++nt)
      for (int r = 0; r < 8; ++r) {
        const float p = __expf(sc[nt][r] - m_i[r]);
        rowsum[r] += p;
        pw[(hsel * 8 + r) * 64 + nt * 16 + lhalf] = (_Float16)p;
      }
    for (int off = 8; off >= 1; off >>= 1)
      for (int r = 0; r < 8; ++r)
        rowsum[r] += __shfl_xor(rowsum[r], off, 16);
    for (int r = 0; r < 8; ++r) l_i[r] = l_i[r] * alpha[r] + rowsum[r];
    for (int nt = 0; nt < 4; ++nt)
      for (int r = 0; r < 8; ++r) o[nt][r] *= alpha[r];

    // ---- O += P * V  (M=16 queries, N=64 dims, K=64 keys) ----
    for (int kk = 0; kk < 64; kk += 32) {
      const _Float16* ap = &pw[lhalf * 64 + kk + hsel * 16];
      FragU ua;
      ua.p.lo = *reinterpret_cast<const v8h*>(ap);
      ua.p.hi = *reinterpret_cast<const v8h*>(ap + 8);
      for (int nt = 0; nt < 4; ++nt) {
        const _Float16* bp = &Vt[(nt * 16 + lhalf) * 64 + kk + hsel * 16];
        v16h bfrag = *reinterpret_cast<const v16h*>(bp);
        o[nt] = WMMA_F16(ua.v, bfrag, o[nt]);
      }
    }
  }

  // Normalize and write [B,S,D] f16 for the output projection GEMM.
  for (int nt = 0; nt < 4; ++nt) {
    const int col = h * NHD + nt * 16 + lhalf;
    for (int r = 0; r < 8; ++r) {
      const int srow = qt * 64 + wave * 16 + hsel * 8 + r;
      const float inv = (l_i[r] > 0.0f) ? (1.0f / l_i[r]) : 0.0f;
      Oh[((size_t)b * NS + srow) * ND + col] = (_Float16)(o[nt][r] * inv);
    }
  }
}

// ---------------------------------------------------------------------------
extern "C" void kernel_launch(void* const* d_in, const int* in_sizes, int n_in,
                              void* d_out, int out_size, void* d_ws, size_t ws_size,
                              hipStream_t stream) {
  const float*          q   = (const float*)d_in[0];
  const unsigned char*  pad = (const unsigned char*)d_in[1];
  const float*          Wq  = (const float*)d_in[2];
  const float*          bq  = (const float*)d_in[3];
  const float*          Wk  = (const float*)d_in[4];
  const float*          bk  = (const float*)d_in[5];
  const float*          Wv  = (const float*)d_in[6];
  const float*          bv  = (const float*)d_in[7];
  const float*          Wo  = (const float*)d_in[8];
  const float*          bo  = (const float*)d_in[9];
  float* out = (float*)d_out;

  const size_t NX = (size_t)NB * NS * ND;   // 4,194,304
  const size_t NW = (size_t)ND * ND;        // 1,048,576

  char* p = (char*)d_ws;
  _Float16* xh  = (_Float16*)p; p += NX * 2;
  _Float16* wqh = (_Float16*)p; p += NW * 2;
  _Float16* wkh = (_Float16*)p; p += NW * 2;
  _Float16* wvh = (_Float16*)p; p += NW * 2;
  _Float16* woh = (_Float16*)p; p += NW * 2;
  _Float16* qh  = (_Float16*)p; p += NX * 2;  // [B,H,S,HD]
  _Float16* kh  = (_Float16*)p; p += NX * 2;
  _Float16* vh  = (_Float16*)p; p += NX * 2;
  _Float16* ah  = (_Float16*)p; p += NX * 2;  // [B,S,D] attention output

  // 1) fp32 -> f16 casts
  cast_f32_to_f16<<<(int)(NX / 1024), 256, 0, stream>>>(q,  xh,  (int)NX);
  cast_f32_to_f16<<<(int)(NW / 1024), 256, 0, stream>>>(Wq, wqh, (int)NW);
  cast_f32_to_f16<<<(int)(NW / 1024), 256, 0, stream>>>(Wk, wkh, (int)NW);
  cast_f32_to_f16<<<(int)(NW / 1024), 256, 0, stream>>>(Wv, wvh, (int)NW);
  cast_f32_to_f16<<<(int)(NW / 1024), 256, 0, stream>>>(Wo, woh, (int)NW);

  // 2) Q/K/V projections (TDM-fed WMMA GEMM, epilogue splits heads)
  dim3 ggrid(32, 8, 1);
  gemm_xwT<<<ggrid, 256, 0, stream>>>(xh, wqh, bq, qh, nullptr, 4096, 0);
  gemm_xwT<<<ggrid, 256, 0, stream>>>(xh, wkh, bk, kh, nullptr, 4096, 0);
  gemm_xwT<<<ggrid, 256, 0, stream>>>(xh, wvh, bv, vh, nullptr, 4096, 0);

  // 3) fused causal+padded flash attention
  dim3 agrid(NS / 64, NB * NH, 1);
  flash_attn<<<agrid, 128, 0, stream>>>(qh, kh, vh, pad, ah);

  // 4) output projection, fp32 result
  gemm_xwT<<<ggrid, 256, 0, stream>>>(ah, woh, bo, nullptr, out, 4096, 1);
}
